// Attention_89120571392370
// MI455X (gfx1250) — matrix operand order repaired and checked by reading, not compiled, not run on hardware
//
#include <hip/hip_runtime.h>
#include <hip/hip_bf16.h>

typedef __attribute__((ext_vector_type(16))) _Float16 v16h;
typedef __attribute__((ext_vector_type(8)))  float    v8f;
typedef __attribute__((ext_vector_type(8)))  _Float16 h8;

#define BQ     2
#define SEQ    2048
#define DIMX   64
#define HEADS  8
#define INNER  4096
#define DHEAD  512
#define SCALEF 0.125f
#define MASKV  (-1e9f)

#define TM  64
#define TN  64
#define TK  32
#define LDK 40   // padded k-stride (f16 elems) -> 80B row stride, keeps 16B alignment

// ---------------- WMMA fragment helpers (wave32, 16x16x32 f16) ----------------
// A (16x32) layout: lane L holds row M=L&15; K-halves selected by L>>4:
//   elems 0..7  -> K = (L>>4)*8 + e        (contiguous 8)
//   elems 8..15 -> K = 16 + (L>>4)*8 + e-8 (contiguous 8)
__device__ __forceinline__ v16h frag_A(const _Float16* As, int row0, int lane) {
  const int r  = row0 + (lane & 15);
  const int kb = (lane >> 4) << 3;
  h8 lo = *(const h8*)(As + r * LDK + kb);
  h8 hi = *(const h8*)(As + r * LDK + kb + 16);
  v16h f;
#pragma unroll
  for (int i = 0; i < 8; ++i) { f[i] = lo[i]; f[i + 8] = hi[i]; }
  return f;
}
// B (32x16) layout: lane L holds col N=L&15; same K order as A (elems 0..7 -> K=(L>>4)*8+e, 8..15 -> K=16+(L>>4)*8+e-8).
// Bt in LDS is stored [n][k] so each lane's 16 K-values sit in two contiguous runs of 8.
__device__ __forceinline__ v16h frag_B(const _Float16* Bt, int n0, int lane) {
  const int r  = n0 + (lane & 15);
  const int kb = (lane >> 4) << 3;
  h8 lo = *(const h8*)(Bt + r * LDK + kb);
  h8 hi = *(const h8*)(Bt + r * LDK + kb + 16);
  v16h f;
#pragma unroll
  for (int i = 0; i < 8; ++i) { f[i] = lo[i]; f[i + 8] = hi[i]; }
  return f;
}

__device__ __forceinline__ v8f wmma16(v16h a, v16h b, v8f c) {
  return __builtin_amdgcn_wmma_f32_16x16x32_f16(false, a, false, b, (short)0, c, false, false);
}

// Async global->LDS B128 copy (CDNA5, ASYNCcnt). GVS addressing:
// dst LDS byte offset (VGPR), 32-bit global byte offset (VGPR), 64-bit SGPR base.
__device__ __forceinline__ void async_copy_b128(const _Float16* lds_dst,
                                                unsigned goff_bytes,
                                                unsigned long long gbase) {
  const unsigned ldsoff = (unsigned)(uintptr_t)lds_dst;  // low 32b = wave LDS offset
  asm volatile("global_load_async_to_lds_b128 %0, %1, %2"
               :
               : "v"(ldsoff), "v"(goff_bytes), "s"(gbase)
               : "memory");
}
__device__ __forceinline__ void wait_asynccnt_le4() {
  asm volatile("s_wait_asynccnt 0x4" ::: "memory");
}
__device__ __forceinline__ void wait_asynccnt_0() {
  asm volatile("s_wait_asynccnt 0x0" ::: "memory");
}

// ---------------- Kernel 1: P = X(f32)[M,Kd] @ W(f32)[Kd,Nn] -> f16 ----------------
__global__ __launch_bounds__(128) void proj_kernel(const float* __restrict__ X,
                                                   const float* __restrict__ W,
                                                   _Float16* __restrict__ Out,
                                                   int Kd, int Nn) {
  __shared__ __align__(16) _Float16 As[TM * LDK];
  __shared__ __align__(16) _Float16 Bt[TN * LDK];
  const int t = threadIdx.x, lane = t & 31, wv = t >> 5;
  const int bm = blockIdx.y * TM, bn = blockIdx.x * TN;
  v8f acc[4];
#pragma unroll
  for (int n = 0; n < 4; ++n)
#pragma unroll
    for (int i = 0; i < 8; ++i) acc[n][i] = 0.f;

  for (int k0 = 0; k0 < Kd; k0 += TK) {
    { // A: 64x32 f32 -> f16 LDS
      const int r = t >> 3, c4 = (t & 7) * 4;
#pragma unroll
      for (int rr = 0; rr < 4; ++rr) {
        const int row = r + rr * 16;
        const float4 v = *(const float4*)(X + (size_t)(bm + row) * Kd + k0 + c4);
        _Float16* d = As + row * LDK + c4;
        d[0] = (_Float16)v.x; d[1] = (_Float16)v.y; d[2] = (_Float16)v.z; d[3] = (_Float16)v.w;
      }
    }
    { // B: W[k][n] 32x64 f32 -> Bt[n][k] f16 (transpose)
      const int kl = t >> 2, ns = (t & 3) * 16;
      const float* src = W + (size_t)(k0 + kl) * Nn + bn + ns;
#pragma unroll
      for (int q = 0; q < 16; q += 4) {
        const float4 v = *(const float4*)(src + q);
        Bt[(ns + q + 0) * LDK + kl] = (_Float16)v.x;
        Bt[(ns + q + 1) * LDK + kl] = (_Float16)v.y;
        Bt[(ns + q + 2) * LDK + kl] = (_Float16)v.z;
        Bt[(ns + q + 3) * LDK + kl] = (_Float16)v.w;
      }
    }
    __syncthreads();
    const v16h a = frag_A(As, wv * 16, lane);
#pragma unroll
    for (int n = 0; n < 4; ++n) acc[n] = wmma16(a, frag_B(Bt, n * 16, lane), acc[n]);
    __syncthreads();
  }
  const int mh = (lane >> 4) << 3, nc = lane & 15;
#pragma unroll
  for (int n = 0; n < 4; ++n)
#pragma unroll
    for (int r = 0; r < 8; ++r)
      Out[(size_t)(bm + wv * 16 + mh + r) * Nn + bn + n * 16 + nc] = (_Float16)acc[n][r];
}

// ---------------- Kernel 2: dots = scale * Q Kt, masked -> f32 ----------------
// Double-buffered: tiles are pure f16 copies, moved with async global->LDS (ASYNCcnt).
__global__ __launch_bounds__(128) void dots_kernel(const _Float16* __restrict__ Q,
                                                   const _Float16* __restrict__ Km,
                                                   const unsigned char* __restrict__ mask,
                                                   float* __restrict__ P) {
  const int bh = blockIdx.z, b = bh / HEADS, h = bh % HEADS;
  const int bm = blockIdx.y * TM, bn = blockIdx.x * TN;
  const int t = threadIdx.x, lane = t & 31, wv = t >> 5;
  __shared__ __align__(16) _Float16 As[2][TM * LDK];
  __shared__ __align__(16) _Float16 Bt[2][TN * LDK];
  v8f acc[4];
#pragma unroll
  for (int n = 0; n < 4; ++n)
#pragma unroll
    for (int i = 0; i < 8; ++i) acc[n][i] = 0.f;

  const size_t ebase = ((size_t)b * SEQ) * INNER + (size_t)h * DHEAD;  // element base
  const unsigned long long qbase = (unsigned long long)(uintptr_t)(Q + ebase);
  const unsigned long long kbase = (unsigned long long)(uintptr_t)(Km + ebase);

  // per-thread tile chunks: ch in {t, t+128}; row = ch>>2 (0..63), off = (ch&3)*8
  const int row0 = t >> 2,            off0 = (t & 3) * 8;
  const int row1 = (t + 128) >> 2,    off1 = ((t + 128) & 3) * 8;

  // issue async copies of one K-step into buffer p
  auto issue = [&](int p, int k0) {
    const unsigned ga0 = (unsigned)(2u * ((unsigned)(bm + row0) * INNER + k0 + off0));
    const unsigned ga1 = (unsigned)(2u * ((unsigned)(bm + row1) * INNER + k0 + off1));
    const unsigned gb0 = (unsigned)(2u * ((unsigned)(bn + row0) * INNER + k0 + off0));
    const unsigned gb1 = (unsigned)(2u * ((unsigned)(bn + row1) * INNER + k0 + off1));
    async_copy_b128(&As[p][row0 * LDK + off0], ga0, qbase);
    async_copy_b128(&As[p][row1 * LDK + off1], ga1, qbase);
    async_copy_b128(&Bt[p][row0 * LDK + off0], gb0, kbase);
    async_copy_b128(&Bt[p][row1 * LDK + off1], gb1, kbase);
  };

  const int nk = DHEAD / TK;   // 16
  issue(0, 0);
  for (int ks = 0; ks < nk; ++ks) {
    const int p = ks & 1;
    if (ks + 1 < nk) {
      issue(p ^ 1, (ks + 1) * TK);
      wait_asynccnt_le4();   // first 4 (current buffer) complete in order
    } else {
      wait_asynccnt_0();
    }
    __syncthreads();         // all waves' copies into buffer p visible
    const v16h a = frag_A(As[p], wv * 16, lane);
#pragma unroll
    for (int n = 0; n < 4; ++n) acc[n] = wmma16(a, frag_B(Bt[p], n * 16, lane), acc[n]);
    __syncthreads();         // everyone done reading p before it is overwritten
  }

  const int mh = (lane >> 4) << 3, nc = lane & 15;
  float* out = P + ((size_t)bh * SEQ) * SEQ;
#pragma unroll
  for (int n = 0; n < 4; ++n) {
    const int j = bn + n * 16 + nc;
    const bool mj = mask[b * SEQ + j] != 0;
#pragma unroll
    for (int r = 0; r < 8; ++r) {
      const int i = bm + wv * 16 + mh + r;
      const bool mi = mask[b * SEQ + i] != 0;
      const float v = acc[n][r] * SCALEF;
      out[(size_t)i * SEQ + j] = (mi && mj) ? v : MASKV;
    }
  }
}

// ---------------- Kernel 3: exact 1.5-entmax per row (len 2048) ----------------
__global__ __launch_bounds__(256) void entmax_kernel(const float* __restrict__ P,
                                                     float* __restrict__ A) {
  const size_t row = blockIdx.x;
  const float* src = P + row * SEQ;
  float* dst = A + row * SEQ;
  __shared__ float z[SEQ];
  __shared__ float s1[SEQ];
  __shared__ float s2[SEQ];
  __shared__ float red1[256];
  __shared__ float red2[256];
  __shared__ int scnt;
  const int t = threadIdx.x, base = t * 8;

  float x[8];
  {
    const float4 va = *(const float4*)(src + base);
    const float4 vb = *(const float4*)(src + base + 4);
    x[0] = va.x * 0.5f; x[1] = va.y * 0.5f; x[2] = va.z * 0.5f; x[3] = va.w * 0.5f;
    x[4] = vb.x * 0.5f; x[5] = vb.y * 0.5f; x[6] = vb.z * 0.5f; x[7] = vb.w * 0.5f;
  }
  float mx = x[0];
#pragma unroll
  for (int i = 1; i < 8; ++i) mx = fmaxf(mx, x[i]);
  red1[t] = mx;
  __syncthreads();
  for (int s = 128; s > 0; s >>= 1) {
    if (t < s) red1[t] = fmaxf(red1[t], red1[t + s]);
    __syncthreads();
  }
  const float rowmax = red1[0];
  __syncthreads();
#pragma unroll
  for (int i = 0; i < 8; ++i) { x[i] -= rowmax; z[base + i] = x[i]; }
  if (t == 0) scnt = 0;
  __syncthreads();

  // bitonic sort, descending
  for (int k = 2; k <= SEQ; k <<= 1) {
    for (int j = k >> 1; j > 0; j >>= 1) {
#pragma unroll
      for (int p = 0; p < 8; ++p) {
        const int i = t + p * 256;
        const int ix = i ^ j;
        if (ix > i) {
          const float a = z[i], b = z[ix];
          const bool desc = ((i & k) == 0);
          if ((a < b) == desc) { z[i] = b; z[ix] = a; }
        }
      }
      __syncthreads();
    }
  }

  // inclusive scan of z and z*z: serial per chunk + block scan of chunk sums
  float a1 = 0.f, a2 = 0.f;
#pragma unroll
  for (int i = 0; i < 8; ++i) {
    const float v = z[base + i];
    a1 += v; a2 += v * v;
    s1[base + i] = a1; s2[base + i] = a2;
  }
  red1[t] = a1; red2[t] = a2;
  __syncthreads();
  for (int off = 1; off < 256; off <<= 1) {
    const float v1 = (t >= off) ? red1[t - off] : 0.f;
    const float v2 = (t >= off) ? red2[t - off] : 0.f;
    __syncthreads();
    red1[t] += v1; red2[t] += v2;
    __syncthreads();
  }
  const float o1 = (t > 0) ? red1[t - 1] : 0.f;
  const float o2 = (t > 0) ? red2[t - 1] : 0.f;

  int c = 0;
#pragma unroll
  for (int i = 0; i < 8; ++i) {
    const float fk = (float)(base + i + 1);
    const float m1 = (s1[base + i] + o1) / fk;
    const float m2 = (s2[base + i] + o2) / fk;
    const float ssv = fk * (m2 - m1 * m1);
    const float delta = (1.0f - ssv) / fk;
    const float tau = m1 - ((delta > 0.f) ? sqrtf(delta) : 0.f);
    const float zv = z[base + i];
    s1[base + i] = tau;            // own slot; safe overwrite
    c += (tau <= zv) ? 1 : 0;
  }
  atomicAdd(&scnt, c);
  __syncthreads();
  const float taus = s1[scnt - 1];
#pragma unroll
  for (int i = 0; i < 8; ++i) {
    float d = x[i] - taus;
    d = fmaxf(d, 0.f);
    dst[base + i] = d * d;
  }
}

// ---------------- Kernel 4: O[b,i,h*D+d] = attn(f32) @ V(f16) ----------------
__global__ __launch_bounds__(128) void av_kernel(const float* __restrict__ At,
                                                 const _Float16* __restrict__ V,
                                                 _Float16* __restrict__ O) {
  const int bh = blockIdx.z, b = bh / HEADS, h = bh % HEADS;
  const int bm = blockIdx.y * TM, bn = blockIdx.x * TN;
  const int t = threadIdx.x, lane = t & 31, wv = t >> 5;
  __shared__ __align__(16) _Float16 As[TM * LDK];
  __shared__ __align__(16) _Float16 Bt[TN * LDK];
  v8f acc[4];
#pragma unroll
  for (int n = 0; n < 4; ++n)
#pragma unroll
    for (int i = 0; i < 8; ++i) acc[n][i] = 0.f;

  const float* Arow = At + (size_t)bh * SEQ * SEQ;
  const size_t vbase = ((size_t)b * SEQ) * INNER + (size_t)h * DHEAD;
  for (int k0 = 0; k0 < SEQ; k0 += TK) {
    { // A: attn f32 -> f16
      const int r = t >> 3, c4 = (t & 7) * 4;
#pragma unroll
      for (int rr = 0; rr < 4; ++rr) {
        const int row = r + rr * 16;
        const float4 v = *(const float4*)(Arow + (size_t)(bm + row) * SEQ + k0 + c4);
        _Float16* d = As + row * LDK + c4;
        d[0] = (_Float16)v.x; d[1] = (_Float16)v.y; d[2] = (_Float16)v.z; d[3] = (_Float16)v.w;
      }
    }
    { // B: V[j][d] f16 -> Bt[d][j] (transpose)
      const int jl = t >> 2, dsg = (t & 3) * 16;
      const _Float16* src = V + vbase + (size_t)(k0 + jl) * INNER + bn + dsg;
      const h8 v0 = *(const h8*)(src);
      const h8 v1 = *(const h8*)(src + 8);
#pragma unroll
      for (int q = 0; q < 8; ++q) {
        Bt[(dsg + q) * LDK + jl]     = v0[q];
        Bt[(dsg + 8 + q) * LDK + jl] = v1[q];
      }
    }
    __syncthreads();
    const v16h a = frag_A(As, wv * 16, lane);
#pragma unroll
    for (int n = 0; n < 4; ++n) acc[n] = wmma16(a, frag_B(Bt, n * 16, lane), acc[n]);
    __syncthreads();
  }
  const int mh = (lane >> 4) << 3, nc = lane & 15;
#pragma unroll
  for (int n = 0; n < 4; ++n)
#pragma unroll
    for (int r = 0; r < 8; ++r)
      O[(size_t)(b * SEQ + bm + wv * 16 + mh + r) * INNER + h * DHEAD + bn + n * 16 + nc] =
          (_Float16)acc[n][r];
}

// ---------------- Kernel 5: Y = O(f16) @ Wo(f32) + bo -> f32 ----------------
__global__ __launch_bounds__(128) void outproj_kernel(const _Float16* __restrict__ O,
                                                      const float* __restrict__ Wo,
                                                      const float* __restrict__ bo,
                                                      float* __restrict__ Y) {
  const int bm = blockIdx.y * TM;
  const int t = threadIdx.x, lane = t & 31, wv = t >> 5;
  __shared__ __align__(16) _Float16 As[TM * LDK];
  __shared__ __align__(16) _Float16 Bt[TN * LDK];
  v8f acc[4];
#pragma unroll
  for (int n = 0; n < 4; ++n)
#pragma unroll
    for (int i = 0; i < 8; ++i) acc[n][i] = 0.f;

  for (int k0 = 0; k0 < INNER; k0 += TK) {
#pragma unroll
    for (int cc = 0; cc < 2; ++cc) { // A: O f16 direct
      const int ch = t + cc * 128, row = ch >> 2, off = (ch & 3) * 8;
      h8 v = *(const h8*)(O + (size_t)(bm + row) * INNER + k0 + off);
      *(h8*)(As + row * LDK + off) = v;
    }
    { // B: Wo[k][n] f32 -> Bt[n][k]
      const int kl = t >> 2, ns = (t & 3) * 16;
      const float* src = Wo + (size_t)(k0 + kl) * DIMX + ns;
#pragma unroll
      for (int q = 0; q < 16; q += 4) {
        const float4 v = *(const float4*)(src + q);
        Bt[(ns + q + 0) * LDK + kl] = (_Float16)v.x;
        Bt[(ns + q + 1) * LDK + kl] = (_Float16)v.y;
        Bt[(ns + q + 2) * LDK + kl] = (_Float16)v.z;
        Bt[(ns + q + 3) * LDK + kl] = (_Float16)v.w;
      }
    }
    __syncthreads();
    const v16h a = frag_A(As, wv * 16, lane);
#pragma unroll
    for (int n = 0; n < 4; ++n) acc[n] = wmma16(a, frag_B(Bt, n * 16, lane), acc[n]);
    __syncthreads();
  }
  const int mh = (lane >> 4) << 3, nc = lane & 15;
#pragma unroll
  for (int n = 0; n < 4; ++n)
#pragma unroll
    for (int r = 0; r < 8; ++r)
      Y[(size_t)(bm + wv * 16 + mh + r) * DIMX + n * 16 + nc] = acc[n][r] + bo[n * 16 + nc];
}

// ---------------- launch ----------------
extern "C" void kernel_launch(void* const* d_in, const int* in_sizes, int n_in,
                              void* d_out, int out_size, void* d_ws, size_t ws_size,
                              hipStream_t stream) {
  (void)in_sizes; (void)n_in; (void)out_size; (void)ws_size;
  const float* x  = (const float*)d_in[0];
  const unsigned char* mask = (const unsigned char*)d_in[1];
  const float* Wq = (const float*)d_in[2];
  const float* Wk = (const float*)d_in[3];
  const float* Wv = (const float*)d_in[4];
  const float* Wo = (const float*)d_in[5];
  const float* bo = (const float*)d_in[6];

  float* outY  = (float*)d_out;                                   // [B,N,DIM]
  float* preA  = outY + (size_t)BQ * SEQ * DIMX;                  // [B,H,N,N]
  float* postA = preA + (size_t)BQ * HEADS * SEQ * SEQ;           // [B,H,N,N]

  _Float16* Qh = (_Float16*)d_ws;
  _Float16* Kh = Qh + (size_t)BQ * SEQ * INNER;
  _Float16* Vh = Kh + (size_t)BQ * SEQ * INNER;
  _Float16* Oh = Vh + (size_t)BQ * SEQ * INNER;

  const dim3 blk(128);
  const dim3 gproj(INNER / TN, (BQ * SEQ) / TM);
  proj_kernel<<<gproj, blk, 0, stream>>>(x, Wq, Qh, DIMX, INNER);
  proj_kernel<<<gproj, blk, 0, stream>>>(x, Wk, Kh, DIMX, INNER);
  proj_kernel<<<gproj, blk, 0, stream>>>(x, Wv, Vh, DIMX, INNER);

  const dim3 gdots(SEQ / TN, SEQ / TM, BQ * HEADS);
  dots_kernel<<<gdots, blk, 0, stream>>>(Qh, Kh, mask, preA);

  entmax_kernel<<<dim3(BQ * HEADS * SEQ), dim3(256), 0, stream>>>(preA, postA);

  const dim3 gav(DHEAD / TN, SEQ / TM, BQ * HEADS);
  av_kernel<<<gav, blk, 0, stream>>>(postA, Vh, Oh);

  const dim3 gout(1, (BQ * SEQ) / TM);
  outproj_kernel<<<gout, blk, 0, stream>>>(Oh, Wo, bo, outY);
}
